// VanillaLSTM_25537875542569
// MI455X (gfx1250) — matrix operand order repaired and checked by reading, not compiled, not run on hardware
//
#include <hip/hip_runtime.h>
#include <hip/hip_bf16.h>
#include <stdint.h>

// ---------------------------------------------------------------------------
// LSTM: B=32, T=1024, I=512, H=1024, gates 4H=4096.
// Phase 1: Xg[t][b][n] = x[b,t,:] @ W_ih^T   (parallel bf16 WMMA GEMM)
// Phase 2 (sequential over t):
//   gates = Xg[t] + h @ W_hh^T   (bf16 WMMA, f32 accum, h staged in LDS,
//                                 global loads double-buffered)
//   pointwise: i,f,g,o -> c,h    (fp32)
// ---------------------------------------------------------------------------

#define LSTM_T 1024
#define LSTM_B 32
#define LSTM_I 512
#define LSTM_H 1024
#define LSTM_G 4096

typedef __attribute__((ext_vector_type(16))) __bf16 v16bf;
typedef __attribute__((ext_vector_type(8)))  float  v8f;

union Frag  { v16bf v; uint4 q[2]; };
union BF16U { unsigned short s; __bf16 b; };

__device__ __forceinline__ __bf16 f2bf(float f) {
  union { float f; unsigned u; } x; x.f = f;
  unsigned r = (x.u + 0x7FFFu + ((x.u >> 16) & 1u)) >> 16;   // RNE
  BF16U o; o.s = (unsigned short)r; return o.b;
}
__device__ __forceinline__ float bf2f(__bf16 b) {
  BF16U i; i.b = b;
  union { unsigned u; float f; } x; x.u = ((unsigned)i.s) << 16; return x.f;
}

// A-matrix fragment 16x32 bf16 (contiguous k order used here, matched by load_frag_b):
// lanes 0-15 : row M=lane,    K = k0+[0..15]
// lanes 16-31: row M=lane-16, K = k0+[16..31]
__device__ __forceinline__ v16bf load_frag_a(const __bf16* row_ptr, int k0, int lane) {
  Frag f;
  const uint4* q = reinterpret_cast<const uint4*>(row_ptr + k0 + ((lane >> 4) << 4));
  f.q[0] = q[0];   // +0..7
  f.q[1] = q[1];   // +8..15
  return f.v;
}

// B-matrix fragment 32x16 bf16:
// lanes 0-15 : col N=lane, K = k0+[0..15];  lanes 16-31: K = k0+[16..31]
__device__ __forceinline__ v16bf load_frag_b(const __bf16* col_ptr, int k0, int lane) {
  Frag f;
  const uint4* q = reinterpret_cast<const uint4*>(col_ptr + k0 + ((lane >> 4) << 4));
  f.q[0] = q[0];
  f.q[1] = q[1];
  return f.v;
}

#define WMMA_BF16(A, B, C) \
  __builtin_amdgcn_wmma_f32_16x16x32_bf16(false, (A), false, (B), (short)0, (C), false, false)

// ------------------------------- helpers -----------------------------------

__global__ void lstm_cvt_bf16(const float* __restrict__ in, __bf16* __restrict__ out, int n) {
  int i = blockIdx.x * blockDim.x + threadIdx.x;
  if (i < n) out[i] = f2bf(in[i]);
}

__global__ void lstm_init_state(__bf16* __restrict__ h, float* __restrict__ c) {
  int i = blockIdx.x * blockDim.x + threadIdx.x;   // 32768 threads
  BF16U z; z.s = 0;
  h[i] = z.b;
  c[i] = 0.0f;
}

// ----------------------- Phase 1: Xg = X @ Wih^T ---------------------------
// M = B*T = 32768 (2048 tiles), N = 4096. Each wave: one M tile x 4 N tiles
// (A fragment reused across the 4 WMMAs per k-step); k-loop double-buffered.
__global__ void __launch_bounds__(256) lstm_xgemm(
    const __bf16* __restrict__ X,    // [32768, 512] row-major (m = b*T + t)
    const __bf16* __restrict__ Wih,  // [4096, 512]  row-major
    __bf16* __restrict__ Xg)         // [T][B][4096] bf16
{
  const int lane = threadIdx.x & 31;
  const int wave = threadIdx.x >> 5;
  const int job  = blockIdx.x * 8 + wave;   // 0..131071
  const int mt   = job >> 6;                // 0..2047
  const int ng   = job & 63;                // 0..63  (64-col group)
  const int m0   = mt << 4;
  const int n0   = ng << 6;
  const int ln   = lane & 15;

  v8f acc0 = {}, acc1 = {}, acc2 = {}, acc3 = {};
  const __bf16* arow = X   + (size_t)(m0 + ln) * LSTM_I;
  const __bf16* b0   = Wih + (size_t)(n0 +  0 + ln) * LSTM_I;
  const __bf16* b1   = Wih + (size_t)(n0 + 16 + ln) * LSTM_I;
  const __bf16* b2   = Wih + (size_t)(n0 + 32 + ln) * LSTM_I;
  const __bf16* b3   = Wih + (size_t)(n0 + 48 + ln) * LSTM_I;

  // prologue: fragments for k = 0
  v16bf a_c  = load_frag_a(arow, 0, lane);
  v16bf w0_c = load_frag_b(b0,   0, lane);
  v16bf w1_c = load_frag_b(b1,   0, lane);
  v16bf w2_c = load_frag_b(b2,   0, lane);
  v16bf w3_c = load_frag_b(b3,   0, lane);

  for (int k = 32; k < LSTM_I; k += 32) {
    // prefetch next k-step before consuming current (overlap loads w/ WMMA)
    v16bf a_n  = load_frag_a(arow, k, lane);
    v16bf w0_n = load_frag_b(b0,   k, lane);
    v16bf w1_n = load_frag_b(b1,   k, lane);
    v16bf w2_n = load_frag_b(b2,   k, lane);
    v16bf w3_n = load_frag_b(b3,   k, lane);

    acc0 = WMMA_BF16(a_c, w0_c, acc0);
    acc1 = WMMA_BF16(a_c, w1_c, acc1);
    acc2 = WMMA_BF16(a_c, w2_c, acc2);
    acc3 = WMMA_BF16(a_c, w3_c, acc3);

    a_c = a_n; w0_c = w0_n; w1_c = w1_n; w2_c = w2_n; w3_c = w3_n;
  }
  // epilogue
  acc0 = WMMA_BF16(a_c, w0_c, acc0);
  acc1 = WMMA_BF16(a_c, w1_c, acc1);
  acc2 = WMMA_BF16(a_c, w2_c, acc2);
  acc3 = WMMA_BF16(a_c, w3_c, acc3);

  // C/D layout: VGPR v -> row m0 + v + 8*(lane>=16), col n + (lane&15)
#pragma unroll
  for (int v = 0; v < 8; ++v) {
    int m  = m0 + v + ((lane >> 4) << 3);
    int tt = m & (LSTM_T - 1);
    int bb = m >> 10;
    size_t base = ((size_t)tt * LSTM_B + bb) * LSTM_G + n0 + ln;
    Xg[base +  0] = f2bf(acc0[v]);
    Xg[base + 16] = f2bf(acc1[v]);
    Xg[base + 32] = f2bf(acc2[v]);
    Xg[base + 48] = f2bf(acc3[v]);
  }
}

// ----------------- Phase 2a: gates = Xg[t] + h @ Whh^T ---------------------
// M = 32 (both tiles per wave -> B-fragment reuse), N = 4096 (256 tiles, one
// per wave). h (64 KB) is staged in LDS once per block; Whh streams from L2
// double-buffered. 32 blocks x 8 waves; 64 KB dynamic LDS per block.
__global__ void __launch_bounds__(256) lstm_hgemm(
    const __bf16* __restrict__ Hb,   // [32, 1024]
    const __bf16* __restrict__ Whh,  // [4096, 1024] row-major
    const __bf16* __restrict__ Xg,   // [T][32][4096]
    float* __restrict__ gates,       // [32, 4096]
    int t)
{
  extern __shared__ char smem[];
  __bf16* hs = (__bf16*)smem;                      // [32][1024] bf16 = 64 KB

  // cooperative stage: 4096 uint4 chunks / 256 threads = 16 each
  {
    const uint4* src = (const uint4*)Hb;
    uint4*       dst = (uint4*)hs;
#pragma unroll
    for (int i = 0; i < 16; ++i)
      dst[threadIdx.x + 256 * i] = src[threadIdx.x + 256 * i];
  }
  __syncthreads();

  const int lane = threadIdx.x & 31;
  const int wave = threadIdx.x >> 5;
  const int nt   = blockIdx.x * 8 + wave;   // 0..255
  const int n0   = nt << 4;
  const int ln   = lane & 15;

  v8f acc0 = {}, acc1 = {};
  const __bf16* a0 = hs  + (size_t)(ln) * LSTM_H;
  const __bf16* a1 = hs  + (size_t)(16 + ln) * LSTM_H;
  const __bf16* br = Whh + (size_t)(n0 + ln) * LSTM_H;

  // prologue
  v16bf w_c  = load_frag_b(br, 0, lane);
  v16bf x0_c = load_frag_a(a0, 0, lane);
  v16bf x1_c = load_frag_a(a1, 0, lane);

  for (int k = 32; k < LSTM_H; k += 32) {
    v16bf w_n  = load_frag_b(br, k, lane);   // global (L2) prefetch
    v16bf x0_n = load_frag_a(a0, k, lane);   // LDS
    v16bf x1_n = load_frag_a(a1, k, lane);   // LDS

    acc0 = WMMA_BF16(x0_c, w_c, acc0);
    acc1 = WMMA_BF16(x1_c, w_c, acc1);

    w_c = w_n; x0_c = x0_n; x1_c = x1_n;
  }
  acc0 = WMMA_BF16(x0_c, w_c, acc0);
  acc1 = WMMA_BF16(x1_c, w_c, acc1);

  const __bf16* xg = Xg + (size_t)t * LSTM_B * LSTM_G;
#pragma unroll
  for (int v = 0; v < 8; ++v) {
    int r   = v + ((lane >> 4) << 3);
    int col = n0 + ln;
    size_t i0 = (size_t)r * LSTM_G + col;
    size_t i1 = (size_t)(r + 16) * LSTM_G + col;
    gates[i0] = acc0[v] + bf2f(xg[i0]);
    gates[i1] = acc1[v] + bf2f(xg[i1]);
  }
}

// --------------------- Phase 2b: pointwise LSTM cell -----------------------
__global__ void lstm_pointwise(
    const float* __restrict__ gates,   // [32, 4096]
    const float* __restrict__ b_ih,    // [4096]
    const float* __restrict__ b_hh,    // [4096]
    float* __restrict__ c,             // [32, 1024] state
    __bf16* __restrict__ Hb,           // [32, 1024] bf16 state for next GEMM
    float* __restrict__ out,           // outputs ++ h_T ++ c_T
    int t)
{
  int idx = blockIdx.x * blockDim.x + threadIdx.x;   // 32768 threads
  int b = idx >> 10;
  int j = idx & (LSTM_H - 1);
  const float* g = gates + (size_t)b * LSTM_G;

  float ig = g[j           ] + b_ih[j           ] + b_hh[j           ];
  float fg = g[j + 1*LSTM_H] + b_ih[j + 1*LSTM_H] + b_hh[j + 1*LSTM_H];
  float gg = g[j + 2*LSTM_H] + b_ih[j + 2*LSTM_H] + b_hh[j + 2*LSTM_H];
  float og = g[j + 3*LSTM_H] + b_ih[j + 3*LSTM_H] + b_hh[j + 3*LSTM_H];

  float i_s = 1.0f / (1.0f + __expf(-ig));
  float f_s = 1.0f / (1.0f + __expf(-fg));
  float g_t = tanhf(gg);
  float o_s = 1.0f / (1.0f + __expf(-og));

  float cn = f_s * c[idx] + i_s * g_t;
  float hn = o_s * tanhf(cn);

  c[idx]  = cn;
  Hb[idx] = f2bf(hn);
  out[((size_t)b * LSTM_T + t) * LSTM_H + j] = hn;
  if (t == LSTM_T - 1) {
    size_t off = (size_t)LSTM_B * LSTM_T * LSTM_H;
    out[off + idx] = hn;                            // h_T
    out[off + (size_t)LSTM_B * LSTM_H + idx] = cn;  // c_T
  }
}

// ---------------------------------------------------------------------------

extern "C" void kernel_launch(void* const* d_in, const int* in_sizes, int n_in,
                              void* d_out, int out_size, void* d_ws, size_t ws_size,
                              hipStream_t stream) {
  (void)in_sizes; (void)n_in; (void)out_size; (void)ws_size;

  const float* x    = (const float*)d_in[0];   // [32, 1024, 512]
  const float* Wih  = (const float*)d_in[1];   // [4096, 512]
  const float* Whh  = (const float*)d_in[2];   // [4096, 1024]
  const float* b_ih = (const float*)d_in[3];   // [4096]
  const float* b_hh = (const float*)d_in[4];   // [4096]
  float* out = (float*)d_out;

  // Workspace layout (256B aligned), total ~315 MB.
  char* ws = (char*)d_ws;
  __bf16* Whh_b = (__bf16*)(ws + 0);            //  8,388,608 B
  __bf16* Wih_b = (__bf16*)(ws + 8388608);      //  4,194,304 B
  __bf16* X_b   = (__bf16*)(ws + 12582912);     // 33,554,432 B
  __bf16* H_b   = (__bf16*)(ws + 46137344);     //     65,536 B
  float*  C_f   = (float*) (ws + 46202880);     //    131,072 B
  float*  G_f   = (float*) (ws + 46333952);     //    524,288 B
  __bf16* Xg_b  = (__bf16*)(ws + 46858240);     // 268,435,456 B

  const int nWhh = LSTM_G * LSTM_H;             // 4,194,304
  const int nWih = LSTM_G * LSTM_I;             // 2,097,152
  const int nX   = LSTM_B * LSTM_T * LSTM_I;    // 16,777,216

  lstm_cvt_bf16<<<(nWhh + 255) / 256, 256, 0, stream>>>(Whh, Whh_b, nWhh);
  lstm_cvt_bf16<<<(nWih + 255) / 256, 256, 0, stream>>>(Wih, Wih_b, nWih);
  lstm_cvt_bf16<<<(nX   + 255) / 256, 256, 0, stream>>>(x,   X_b,   nX);
  lstm_init_state<<<128, 256, 0, stream>>>(H_b, C_f);

  // Phase 1: 2048 M-tiles x 64 N-groups = 131072 wave-jobs / 8 per block.
  lstm_xgemm<<<16384, 256, 0, stream>>>(X_b, Wih_b, Xg_b);

  // Phase 2: sequential recurrence; kernel boundaries provide global sync.
  const size_t lds_bytes = (size_t)LSTM_B * LSTM_H * sizeof(__bf16);  // 64 KB
  for (int t = 0; t < LSTM_T; ++t) {
    lstm_hgemm<<<32, 256, lds_bytes, stream>>>(H_b, Whh_b, Xg_b, G_f, t);
    lstm_pointwise<<<128, 256, 0, stream>>>(G_f, b_ih, b_hh, C_f, H_b, out, t);
  }
}
